// Mutual_Information_Loss_2645699854871
// MI455X (gfx1250) — compile-verified, matrix operand order checked by reading, not run on hardware
//
#include <hip/hip_runtime.h>
#include <hip/hip_bf16.h>

typedef __attribute__((ext_vector_type(16))) _Float16 v16h;
typedef __attribute__((ext_vector_type(8)))  float    v8f;

// ---------------------------------------------------------------------------
// Kernel: zero a u32/f32 buffer (grid-stride)
// ---------------------------------------------------------------------------
__global__ void mi_zero_u32(unsigned* __restrict__ p, unsigned n) {
    unsigned i = blockIdx.x * blockDim.x + threadIdx.x;
    unsigned stride = gridDim.x * blockDim.x;
    for (; i < n; i += stride) p[i] = 0u;
}

// ---------------------------------------------------------------------------
// Kernel: fused L2-normalize (over channel dim, 64) + integer histogram.
// x: [8,64,256,256] f32.  hist: [W=256][H=256][256] u32, counts over (b,c).
// One thread per (b, w, h/4); float4 vector loads, two strided channel passes
// (second pass hits L2: one input slice is 134 MB < 192 MB L2).
// ---------------------------------------------------------------------------
__device__ __forceinline__ void mi_bump(float v, unsigned* __restrict__ h) {
    float fv = floorf(v);
    if (v == fv && v >= 0.0f && v <= 255.0f) {
        atomicAdd(h + (int)fv, 1u);
    }
}

__global__ void mi_norm_hist(const float* __restrict__ x,
                             unsigned* __restrict__ hist) {
    int tid = blockIdx.x * blockDim.x + threadIdx.x;   // 131072 threads
    int h4 = tid & 63;           // 64 groups of 4 along h
    int w  = (tid >> 6) & 255;
    int b  = tid >> 14;          // 0..7
    // element index of (b, c=0, w, h4*4)
    unsigned base = ((unsigned)(b * 64) * 256u + (unsigned)w) * 256u + (unsigned)(h4 * 4);

    float4 s = make_float4(0.f, 0.f, 0.f, 0.f);
    #pragma unroll 4
    for (int c = 0; c < 64; ++c) {
        float4 v = *(const float4*)(x + base + (unsigned)c * 65536u);
        s.x += v.x * v.x; s.y += v.y * v.y;
        s.z += v.z * v.z; s.w += v.w * v.w;
    }
    float4 inv;
    inv.x = 1.0f / fmaxf(sqrtf(s.x), 1e-12f);
    inv.y = 1.0f / fmaxf(sqrtf(s.y), 1e-12f);
    inv.z = 1.0f / fmaxf(sqrtf(s.z), 1e-12f);
    inv.w = 1.0f / fmaxf(sqrtf(s.w), 1e-12f);

    unsigned hbase = (unsigned)w * 65536u + (unsigned)(h4 * 4) * 256u;
    #pragma unroll 4
    for (int c = 0; c < 64; ++c) {
        float4 v = *(const float4*)(x + base + (unsigned)c * 65536u);
        mi_bump(v.x * inv.x, hist + hbase + 0u * 256u);
        mi_bump(v.y * inv.y, hist + hbase + 1u * 256u);
        mi_bump(v.z * inv.z, hist + hbase + 2u * 256u);
        mi_bump(v.w * inv.w, hist + hbase + 3u * 256u);
    }
}

// ---------------------------------------------------------------------------
// Kernel: entropy over h:  e[w,i] = -sum_h p*log(p+1e-8), p = hist/65536
// grid = 256 (w), block = 256 (i); loads coalesced across i.
// ---------------------------------------------------------------------------
__global__ void mi_entropy(const unsigned* __restrict__ hist,
                           float* __restrict__ e) {
    int w = blockIdx.x, i = threadIdx.x;
    const unsigned* p = hist + (unsigned)w * 65536u + (unsigned)i;
    float acc = 0.0f;
    #pragma unroll 4
    for (int h = 0; h < 256; ++h) {
        float pv = (float)p[h * 256] * (1.0f / 65536.0f);
        acc += pv * __logf(pv + 1e-8f);
    }
    e[w * 256 + i] = -acc;
}

// ---------------------------------------------------------------------------
// Kernel: joint entropy via WMMA.  Per batch b (block):
//   M[c][i] = (e_ms[b,c]==i), P[c][j] = (e_p[b,c]==j)   (0/1, exact in f16)
//   s[i,j]  = 2*(M^T P)[i,j] + 256 - rm[i] - rp[j]
//   joint[b,j] += -256 * sum_i pm*log(pm+1e-8),  pm = s/65536
// Masks stored K-major in LDS so each lane's A/B fragment is two contiguous
// 16-byte chunks (ds_load_b128).  8 waves x 32 tiles = 256 16x16 tiles.
// Dynamic LDS: 2*128KB masks + 4KB aux = 266240 B (<320KB/WGP on CDNA5).
// ---------------------------------------------------------------------------
__global__ void mi_joint_wmma(const float* __restrict__ e_p,   // x_p  (fo entropy)
                              const float* __restrict__ e_ms,  // x_ms (f5 entropy)
                              float* __restrict__ joint) {
    extern __shared__ unsigned char smem[];
    _Float16* Mt = (_Float16*)smem;                  // [i][c]  (A = M^T, K-major)
    _Float16* Pt = (_Float16*)(smem + 131072);       // [j][c]  (B,      K-major)
    float* ems = (float*)(smem + 262144);
    float* ep  = ems + 256;
    float* rm  = ep  + 256;
    float* rp  = rm  + 256;

    const int b = blockIdx.x;
    const int t = threadIdx.x;

    ems[t] = e_ms[b * 256 + t];
    ep[t]  = e_p[b * 256 + t];
    __syncthreads();

    // build one-hot mask matrices (transposed: row = bin index, col = c)
    for (int n = t; n < 65536; n += 256) {
        int i = n >> 8, c = n & 255;
        float fi = (float)i;
        Mt[n] = (ems[c] == fi) ? (_Float16)1.0f : (_Float16)0.0f;
        Pt[n] = (ep[c]  == fi) ? (_Float16)1.0f : (_Float16)0.0f;
    }
    __syncthreads();

    // column sums over c (rows of the transposed layout are contiguous)
    {
        float sm = 0.0f, sp = 0.0f;
        const _Float16* mrow = Mt + t * 256;
        const _Float16* prow = Pt + t * 256;
        #pragma unroll 8
        for (int c = 0; c < 256; ++c) { sm += (float)mrow[c]; sp += (float)prow[c]; }
        rm[t] = sm; rp[t] = sp;
    }
    __syncthreads();

    const int wave = t >> 5;
    const int lane = t & 31;
    const int lo16 = lane & 15;
    const int sel  = (lane >= 16) ? 8 : 0;   // K-half select per WMMA A/B layout

    for (int tile = wave * 32; tile < wave * 32 + 32; ++tile) {
        const int ti = tile >> 4;
        const int tj = tile & 15;
        v8f acc = {0.f, 0.f, 0.f, 0.f, 0.f, 0.f, 0.f, 0.f};

        const _Float16* arow = Mt + (ti * 16 + lo16) * 256;
        const _Float16* brow = Pt + (tj * 16 + lo16) * 256;

        #pragma unroll
        for (int k0 = 0; k0 < 256; k0 += 32) {
            const _Float16* pa = arow + k0 + sel;
            const _Float16* pb = brow + k0 + sel;
            v16h a, bm;
            #pragma unroll
            for (int q = 0; q < 8; ++q) {
                a[q]      = pa[q];
                a[8 + q]  = pa[16 + q];
                bm[q]     = pb[q];
                bm[8 + q] = pb[16 + q];
            }
            acc = __builtin_amdgcn_wmma_f32_16x16x32_f16(
                    false, a, false, bm, (short)0, acc, false, false);
        }

        // epilogue: identity + entropy term, reduce over the 8 i-rows per lane
        const int jG  = tj * 16 + lo16;
        const float rpj = rp[jG];
        float local = 0.0f;
        #pragma unroll
        for (int r = 0; r < 8; ++r) {
            const int iG = ti * 16 + r + sel;            // C/D layout: M = r (+8 hi half)
            float s  = 2.0f * acc[r] + 256.0f - rm[iG] - rpj;
            float pm = s * (1.0f / 65536.0f);
            local += pm * __logf(pm + 1e-8f);
        }
        atomicAdd(&joint[b * 256 + jG], -256.0f * local);
    }
}

// ---------------------------------------------------------------------------
// Kernel: smooth-L1 mean over [256,256] -> scalar
// ---------------------------------------------------------------------------
__global__ void mi_loss(const float* __restrict__ e1, const float* __restrict__ e2,
                        const float* __restrict__ joint, float* __restrict__ out) {
    __shared__ float red[256];
    int t = threadIdx.x;
    float acc = 0.0f;
    for (int n = t; n < 65536; n += 256) {
        float a  = e1[n] + e2[n];
        float d  = a - joint[n];
        float ad = fabsf(d);
        acc += (ad < 1.0f) ? 0.5f * d * d : ad - 0.5f;
    }
    red[t] = acc;
    __syncthreads();
    for (int s = 128; s > 0; s >>= 1) {
        if (t < s) red[t] += red[t + s];
        __syncthreads();
    }
    if (t == 0) out[0] = red[0] * (1.0f / 65536.0f);
}

// ---------------------------------------------------------------------------
extern "C" void kernel_launch(void* const* d_in, const int* in_sizes, int n_in,
                              void* d_out, int out_size, void* d_ws, size_t ws_size,
                              hipStream_t stream) {
    (void)in_sizes; (void)n_in; (void)out_size; (void)ws_size;
    const float* fo = (const float*)d_in[0];   // feature_output [8,64,256,256]
    const float* f5 = (const float*)d_in[1];   // f_5            [8,64,256,256]
    float* out = (float*)d_out;                // scalar

    // workspace: hist (64MB) | e1 | e2 | joint
    unsigned* hist = (unsigned*)d_ws;
    float* e1    = (float*)((char*)d_ws + 67108864);
    float* e2    = e1 + 65536;
    float* joint = e2 + 65536;

    const unsigned histN = 256u * 256u * 256u;   // 16,777,216

    // --- input 1: normalize + histogram + entropy ---
    mi_zero_u32<<<4096, 256, 0, stream>>>(hist, histN);
    mi_norm_hist<<<512, 256, 0, stream>>>(fo, hist);
    mi_entropy<<<256, 256, 0, stream>>>(hist, e1);

    // --- input 2: reuse hist ---
    mi_zero_u32<<<4096, 256, 0, stream>>>(hist, histN);
    mi_norm_hist<<<512, 256, 0, stream>>>(f5, hist);
    mi_entropy<<<256, 256, 0, stream>>>(hist, e2);

    // --- joint entropy: 256 batched 256x256x256 GEMMs on the WMMA path ---
    mi_zero_u32<<<256, 256, 0, stream>>>((unsigned*)joint, 65536u);
    mi_joint_wmma<<<256, 256, 266240, stream>>>(/*e_p=*/e1, /*e_ms=*/e2, joint);

    // --- smooth-L1 mean -> scalar ---
    mi_loss<<<1, 256, 0, stream>>>(e1, e2, joint, out);
}
